// StructuralAttentionLayer_46669114638374
// MI455X (gfx1250) — matrix operand order repaired and checked by reading, not compiled, not run on hardware
//
#include <hip/hip_runtime.h>
#include <hip/hip_bf16.h>

typedef __attribute__((ext_vector_type(16))) _Float16 v16h;
typedef __attribute__((ext_vector_type(8)))  float    v8f;
typedef __attribute__((ext_vector_type(4)))  float    v4f;

#define H_HEADS 8
#define D_IN    128
#define D_HEAD  16
#define D_OUT   128   // H_HEADS * D_HEAD

// ---------------------------------------------------------------------------
// Kernel 0: pre-swizzle W [H,128,16] fp32 -> per-lane B-fragment table (f16).
// B (32x16, K x N) VGPR layout (wave32): lanes 0-15 hold K=0..7,16..23 (N=lane),
// lanes 16-31 hold K=8..15,24..31; VGPR r packs K pair (2r, 2r+1) lo/hi half.
// Table layout: Wt[((h*4 + kc)*32 + lane)*16 + j] = W[h][kc*32 + (lane>>4)*8 + j + (j>=8 ? 8 : 0)][lane&15]
// ---------------------------------------------------------------------------
__global__ void prep_wt_kernel(const float* __restrict__ W, _Float16* __restrict__ Wt) {
    int t = blockIdx.x * blockDim.x + threadIdx.x;   // 0 .. H*4*32-1
    if (t >= H_HEADS * 4 * 32) return;
    int lane = t & 31;
    int kc   = (t >> 5) & 3;
    int h    = t >> 7;
    int nn    = lane & 15;
    int kbase = kc * 32 + (lane >> 4) * 8;
    _Float16* dstp = Wt + (size_t)t * 16;
    #pragma unroll
    for (int j = 0; j < 16; ++j) {
        dstp[j] = (_Float16)W[(size_t)h * (D_IN * D_HEAD) + (size_t)(kbase + j + (j >= 8 ? 8 : 0)) * D_HEAD + nn];
    }
}

// ---------------------------------------------------------------------------
// Kernel 1: feats[n, h*16+o] = sum_d x[n,d] * W[h,d,o]  via v_wmma_f32_16x16x32_f16
// One wave computes a 16-row x 128-col tile: 4 K-chunks x 8 head tiles.
// A layout (16-bit A 16x32): lanes0-15: M=lane, K = {0..7} in v0-3, {16..23} in v4-7;
// lanes16-31: K = {8..15}, {24..31}  -> two contiguous 8-element runs per lane.
// ---------------------------------------------------------------------------
__global__ void gemm_feats_kernel(const float* __restrict__ x,
                                  const _Float16* __restrict__ Wt,
                                  float* __restrict__ feats, int n_nodes) {
    int wave = (int)((blockIdx.x * blockDim.x + threadIdx.x) >> 5);
    int lane = threadIdx.x & 31;
    int m0 = wave * 16;
    if (m0 >= n_nodes) return;

    int row = m0 + (lane & 15);
    if (row >= n_nodes) row = n_nodes - 1;           // tail clamp (loads only)
    int seg = lane >> 4;

    v8f acc[H_HEADS];
    #pragma unroll
    for (int h = 0; h < H_HEADS; ++h)
        acc[h] = (v8f){0.f, 0.f, 0.f, 0.f, 0.f, 0.f, 0.f, 0.f};

    #pragma unroll
    for (int kc = 0; kc < 4; ++kc) {
        const float* xp0 = x + (size_t)row * D_IN + kc * 32 + seg * 8;
        const float* xp1 = xp0 + 16;
        v4f p00 = *(const v4f*)(xp0);
        v4f p01 = *(const v4f*)(xp0 + 4);
        v4f p10 = *(const v4f*)(xp1);
        v4f p11 = *(const v4f*)(xp1 + 4);
        v16h a;
        #pragma unroll
        for (int j = 0; j < 4; ++j) {
            a[j]      = (_Float16)p00[j];
            a[4 + j]  = (_Float16)p01[j];
            a[8 + j]  = (_Float16)p10[j];
            a[12 + j] = (_Float16)p11[j];
        }
        #pragma unroll
        for (int h = 0; h < H_HEADS; ++h) {
            v16h b = *(const v16h*)(Wt + (((size_t)h * 4 + kc) * 32 + lane) * 16);
            acc[h] = __builtin_amdgcn_wmma_f32_16x16x32_f16(
                false, a, false, b, (short)0, acc[h], false, false);
        }
    }

    // C/D layout: VGPR r -> lanes0-15: (M=r, N=lane); lanes16-31: (M=8+r, N=lane-16)
    int nn   = lane & 15;
    int mofs = (lane >> 4) * 8;
    #pragma unroll
    for (int h = 0; h < H_HEADS; ++h) {
        #pragma unroll
        for (int r = 0; r < 8; ++r) {
            int m = m0 + mofs + r;
            if (m < n_nodes)
                feats[(size_t)m * D_OUT + h * D_HEAD + nn] = acc[h][r];
        }
    }
}

// ---------------------------------------------------------------------------
// Kernel 2: f1[n,h] = feats[n, h*16:] . a1[h] + b1[h]   (same for f2)
// ---------------------------------------------------------------------------
__global__ void node_scalars_kernel(const float* __restrict__ feats,
                                    const float* __restrict__ a1, const float* __restrict__ b1,
                                    const float* __restrict__ a2, const float* __restrict__ b2,
                                    float* __restrict__ f1, float* __restrict__ f2, int n_nodes) {
    int t = blockIdx.x * blockDim.x + threadIdx.x;
    if (t >= n_nodes * H_HEADS) return;
    int h = t & 7;
    int n = t >> 3;
    const float* fr = feats + (size_t)n * D_OUT + h * D_HEAD;
    float s1 = 0.f, s2 = 0.f;
    #pragma unroll
    for (int o = 0; o < D_HEAD; ++o) {
        float v = fr[o];
        s1 += v * a1[h * D_HEAD + o];
        s2 += v * a2[h * D_HEAD + o];
    }
    f1[t] = s1 + b1[h];
    f2[t] = s2 + b2[h];
}

// ---------------------------------------------------------------------------
// Kernel 3: row_ptr from sorted src (lower_bound per node)
// ---------------------------------------------------------------------------
__global__ void build_rowptr_kernel(const int* __restrict__ src, int* __restrict__ row_ptr,
                                    int n_nodes, int n_edges) {
    int i = blockIdx.x * blockDim.x + threadIdx.x;
    if (i > n_nodes) return;
    int lo = 0, hi = n_edges;
    while (lo < hi) {
        int mid = (lo + hi) >> 1;
        if (src[mid] < i) lo = mid + 1; else hi = mid;
    }
    row_ptr[i] = lo;
}

// ---------------------------------------------------------------------------
// Kernel 4: one wave per node. Pass1: edge-parallel max of leaky-relu logits.
// Pass2: fused exp / denom-sum / weighted feature accumulation (no stored
// coef, no stored logits). Then normalize + ELU + store 16B/lane.
// Lane owns channels c = lane*4..lane*4+3, head h = lane>>2.
// ---------------------------------------------------------------------------
__global__ void gat_aggregate_kernel(const float* __restrict__ feats,
                                     const float* __restrict__ f1,
                                     const float* __restrict__ f2,
                                     const int* __restrict__ dstI,
                                     const int* __restrict__ row_ptr,
                                     float* __restrict__ out, int n_nodes) {
    int wave = (int)((blockIdx.x * blockDim.x + threadIdx.x) >> 5);
    int lane = threadIdx.x & 31;
    if (wave >= n_nodes) return;
    int i  = wave;
    int r0 = row_ptr[i];
    int r1 = row_ptr[i + 1];
    int c  = lane * 4;
    int h  = lane >> 2;

    if (r0 == r1) {                       // no out-edges: segment_sum = 0, elu(0)=0
        v4f z = {0.f, 0.f, 0.f, 0.f};
        *(v4f*)(out + (size_t)i * D_OUT + c) = z;
        return;
    }

    v4f f1a = *(const v4f*)(f1 + (size_t)i * H_HEADS);
    v4f f1b = *(const v4f*)(f1 + (size_t)i * H_HEADS + 4);
    float f1v[H_HEADS] = {f1a[0], f1a[1], f1a[2], f1a[3], f1b[0], f1b[1], f1b[2], f1b[3]};

    // ---- pass 1: per-head segment max (edges striped across lanes) ----
    float mx[H_HEADS];
    #pragma unroll
    for (int k = 0; k < H_HEADS; ++k) mx[k] = -3.4e38f;
    for (int e = r0 + lane; e < r1; e += 32) {
        int d = dstI[e];
        v4f g0 = *(const v4f*)(f2 + (size_t)d * H_HEADS);
        v4f g1 = *(const v4f*)(f2 + (size_t)d * H_HEADS + 4);
        float gv[H_HEADS] = {g0[0], g0[1], g0[2], g0[3], g1[0], g1[1], g1[2], g1[3]};
        #pragma unroll
        for (int k = 0; k < H_HEADS; ++k) {
            float t = f1v[k] + gv[k];
            t = fmaxf(0.2f * t, t);       // leaky relu
            mx[k] = fmaxf(mx[k], t);
        }
    }
    #pragma unroll
    for (int k = 0; k < H_HEADS; ++k) {
        #pragma unroll
        for (int m = 1; m < 32; m <<= 1)
            mx[k] = fmaxf(mx[k], __shfl_xor(mx[k], m, 32));
    }

    // ---- pass 2: exp, denom, weighted accumulation (lanes cooperate per edge) ----
    float mh  = mx[h];
    float f1h = f1v[h];
    v4f acc = {0.f, 0.f, 0.f, 0.f};
    float dsum = 0.f;
    for (int e = r0; e < r1; ++e) {
        int d = dstI[e];
        float w = f1h + f2[(size_t)d * H_HEADS + h];
        w = fmaxf(0.2f * w, w);
        float ex = __expf(w - mh);
        dsum += ex;
        v4f fv = *(const v4f*)(feats + (size_t)d * D_OUT + c);
        acc += ex * fv;
    }

    float inv = 1.0f / dsum;
    v4f res;
    #pragma unroll
    for (int j = 0; j < 4; ++j) {
        float v = acc[j] * inv;
        res[j] = (v > 0.f) ? v : (__expf(v) - 1.0f);   // ELU (alpha=1)
    }
    *(v4f*)(out + (size_t)i * D_OUT + c) = res;
}

// ---------------------------------------------------------------------------
extern "C" void kernel_launch(void* const* d_in, const int* in_sizes, int n_in,
                              void* d_out, int out_size, void* d_ws, size_t ws_size,
                              hipStream_t stream) {
    const float* x  = (const float*)d_in[0];
    const float* W  = (const float*)d_in[1];
    const float* a1 = (const float*)d_in[2];
    const float* b1 = (const float*)d_in[3];
    const float* a2 = (const float*)d_in[4];
    const float* b2 = (const float*)d_in[5];
    const int*   src = (const int*)d_in[6];
    const int*   dst = (const int*)d_in[7];

    int n = in_sizes[0] / D_IN;    // nodes
    int e = in_sizes[6];           // edges
    (void)n_in; (void)out_size; (void)ws_size;

    // workspace carve-up (256B aligned slabs)
    char* ws = (char*)d_ws;
    size_t off = 0;
    auto carve = [&](size_t bytes) -> void* {
        void* p = ws + off;
        off = (off + bytes + 255) & ~(size_t)255;
        return p;
    };
    float*     feats   = (float*)carve((size_t)n * D_OUT * sizeof(float));
    float*     f1      = (float*)carve((size_t)n * H_HEADS * sizeof(float));
    float*     f2      = (float*)carve((size_t)n * H_HEADS * sizeof(float));
    int*       row_ptr = (int*)carve((size_t)(n + 1) * sizeof(int));
    _Float16*  Wt      = (_Float16*)carve((size_t)H_HEADS * 4 * 32 * 16 * sizeof(_Float16));

    // 0) swizzle W into WMMA B-fragment layout (f16)
    prep_wt_kernel<<<4, 256, 0, stream>>>(W, Wt);

    // 1) feats = x @ W  (WMMA, one wave per 16-row tile)
    int ntiles = (n + 15) / 16;
    int gblocks = (ntiles * 32 + 255) / 256;
    gemm_feats_kernel<<<gblocks, 256, 0, stream>>>(x, Wt, feats, n);

    // 2) per-node attention scalars
    node_scalars_kernel<<<(n * H_HEADS + 255) / 256, 256, 0, stream>>>(
        feats, a1, b1, a2, b2, f1, f2, n);

    // 3) CSR row pointers from sorted src
    build_rowptr_kernel<<<(n + 1 + 255) / 256, 256, 0, stream>>>(src, row_ptr, n, e);

    // 4) fused segment-softmax + SPMM + ELU (one wave per node)
    int ablocks = ((size_t)n * 32 + 255) / 256;
    gat_aggregate_kernel<<<ablocks, 256, 0, stream>>>(
        feats, f1, f2, dst, row_ptr, (float*)d_out, n);
}
